// CEKT_58437325029818
// MI455X (gfx1250) — matrix operand order repaired and checked by reading, not compiled, not run on hardware
//
#include <hip/hip_runtime.h>
#include <hip/hip_bf16.h>
#include <math.h>

typedef unsigned short u16;
typedef __bf16 bf16_t;
typedef __attribute__((ext_vector_type(16))) bf16_t v16bf;
typedef __attribute__((ext_vector_type(8)))  float  v8f;
typedef __attribute__((ext_vector_type(8)))  unsigned v8u;

#define TT   200   // T
#define TS   199   // T-1
#define BB   256   // batch
#define KC   256   // NUM_KC
#define DH   128   // D_H == D_E == D_K == D_KE
#define NH   4
#define NEX  10000

// ---------- bf16 helpers ----------
__device__ __forceinline__ u16 f2bf(float f){
  unsigned u = __builtin_bit_cast(unsigned, f);
  u += 0x7FFFu + ((u >> 16) & 1u);
  return (u16)(u >> 16);
}
__device__ __forceinline__ float bf2f(u16 h){
  unsigned u = ((unsigned)h) << 16;
  return __builtin_bit_cast(float, u);
}
__device__ __forceinline__ void zero8(v8f& c){
  #pragma unroll
  for(int r=0;r<8;++r) c[r]=0.f;
}

// ---------- fragment loads: two 8-half runs (2 x b128 each) ----------
__device__ __forceinline__ v16bf pack2(uint4 c0, uint4 c1){
  v8u u;
  u[0]=c0.x; u[1]=c0.y; u[2]=c0.z; u[3]=c0.w;
  u[4]=c1.x; u[5]=c1.y; u[6]=c1.z; u[7]=c1.w;
  return __builtin_bit_cast(v16bf, u);
}
// A-matrix fragment from row-major [M][K] bf16: per ISA layout, lane's 16 halfs are
// two contiguous 8-half runs: K = (hi?8:0)+0..7 and K = (hi?24:16)+0..7
__device__ __forceinline__ v16bf load_frag_rm(const u16* m, int row0, int ld, int k0, int lane){
  int l = lane & 15, hi = lane >> 4;
  const u16* p = m + (size_t)(row0 + l) * ld + k0 + (hi ? 8 : 0);
  uint4 c0 = *reinterpret_cast<const uint4*>(p);
  uint4 c1 = *reinterpret_cast<const uint4*>(p + 16);
  return pack2(c0, c1);
}
// B-matrix fragment from transposed storage [N][K] bf16: lane n0+l reads K = (hi?8:0)+0..7 and K = (hi?24:16)+0..7
__device__ __forceinline__ v16bf load_frag_bt(const u16* mt, int n0, int ldk, int k0, int lane){
  int l = lane & 15, hi = lane >> 4;
  const u16* p = mt + (size_t)(n0 + l) * ldk + k0 + (hi ? 8 : 0);
  uint4 c0 = *reinterpret_cast<const uint4*>(p);
  uint4 c1 = *reinterpret_cast<const uint4*>(p + 16);
  return pack2(c0, c1);
}
__device__ __forceinline__ v8f wmma_bf(v16bf a, v16bf b, v8f c){
  return __builtin_amdgcn_wmma_f32_16x16x32_bf16(false, a, false, b, (short)0, c, false, false);
}
// f32 C/D 16x16: VGPR r -> (row0 + r + (lane>=16?8:0), n0 + (lane&15))
__device__ __forceinline__ void store_d_bf(u16* m, v8f d, int row0, int ld, int n0, int lane){
  int l = lane & 15, mo = (lane >> 4) ? 8 : 0;
  #pragma unroll
  for(int r=0;r<8;++r) m[(size_t)(row0 + r + mo) * ld + n0 + l] = f2bf(d[r]);
}

// K offset for A fragment half index (used by on-the-fly attention fragments)
__device__ __forceinline__ int a_koff(int i, int hi){
  int vg = i >> 1, o = i & 1;
  return (vg < 4) ? ((hi ? 8 : 0) + 2*vg + o)
                  : ((hi ? 24 : 16) + 2*(vg-4) + o);
}
// masked leaky-relu softmax attention fragment, generated in registers
__device__ __forceinline__ v16bf att_frag(const float* f1, const float* f2,
                                          const float* rmax, const float* rinv,
                                          const unsigned* mb, int row0, int k0, int lane){
  int l = lane & 15, hi = lane >> 4;
  int r = row0 + l;
  float F1 = f1[r], M = rmax[r], I = rinv[r];
  v16bf a;
  #pragma unroll
  for(int i=0;i<16;++i){
    int j = k0 + a_koff(i, hi);
    float s = F1 + f2[j];
    s = (s > 0.f) ? s : 0.2f*s;
    bool mk = (mb[r*8 + (j>>5)] >> (j & 31)) & 1u;
    float v = mk ? (__expf(s - M) * I) : 0.f;
    a[i] = __builtin_bit_cast(bf16_t, f2bf(v));
  }
  return a;
}

// ===================== weight conversion (transpose to [N][K] bf16) =====================
__global__ void cvt_bf_t_k(const float* __restrict__ src, u16* __restrict__ dst, int n){
  // n = rows*cols with rows==cols==128: dst[c*128+r] = bf(src[r*128+c])
  int i = blockIdx.x*256 + threadIdx.x;
  if(i < n){
    int r = i >> 7, c = i & 127;
    dst[(size_t)c*128 + r] = f2bf(src[(size_t)r*128 + c]);
  }
}

// rg/rhc = resp_emb @ Wg/Whc[256:320] + bias   (2 x 128)
__global__ void resp_fold_k(const float* __restrict__ resp, const float* __restrict__ Wg,
                            const float* __restrict__ bg, const float* __restrict__ Whc,
                            const float* __restrict__ bhc, float* __restrict__ rg,
                            float* __restrict__ rhc){
  int t = threadIdx.x; int j = t >> 7, d = t & 127;
  float sg = bg[d], sc = bhc[d];
  for(int k=0;k<64;++k){
    float r = resp[j*64 + k];
    sg += r * Wg [(256 + k)*DH + d];
    sc += r * Whc[(256 + k)*DH + d];
  }
  rg[t] = sg; rhc[t] = sc;
}

// eg[q]=ex_emb[q]@Wg[128:256], ehc[q]=ex_emb[q]@Whc[128:256], ydot[q]=ex_emb[q].Wp[128:256]
__global__ void __launch_bounds__(256) qproj_k(const float* __restrict__ ex_emb,
                        const u16* __restrict__ wg1t, const u16* __restrict__ whc1t,
                        const float* __restrict__ Wp, float* __restrict__ eg,
                        float* __restrict__ ehc, float* __restrict__ ydot){
  __shared__ u16 sE[256*DH];
  int tid = threadIdx.x, lane = tid & 31, wv = tid >> 5;
  int q0 = blockIdx.x * 256;
  int q = q0 + tid; bool ok = q < NEX;
  const float* er = ex_emb + (size_t)(ok ? q : 0) * DH;
  for(int d=lane*4; d<DH; d+=0x7fffffff){ break; }   // (placeholder removed below)
  for(int d=0; d<DH; d+=4){
    float4 v = *reinterpret_cast<const float4*>(er + d);
    u16* o = sE + tid*DH + d;
    o[0]=f2bf(ok?v.x:0.f); o[1]=f2bf(ok?v.y:0.f); o[2]=f2bf(ok?v.z:0.f); o[3]=f2bf(ok?v.w:0.f);
  }
  __syncthreads();
  int row0 = wv * 32;
  v8f cg[16], cc[16];
  #pragma unroll
  for(int i=0;i<16;++i){ zero8(cg[i]); zero8(cc[i]); }
  #pragma unroll
  for(int kt=0; kt<4; ++kt){
    v16bf a0 = load_frag_rm(sE, row0,    DH, kt*32, lane);
    v16bf a1 = load_frag_rm(sE, row0+16, DH, kt*32, lane);
    #pragma unroll
    for(int nt=0; nt<8; ++nt){
      v16bf b1 = load_frag_bt(wg1t,  nt*16, DH, kt*32, lane);
      cg[nt]   = wmma_bf(a0, b1, cg[nt]);
      cg[8+nt] = wmma_bf(a1, b1, cg[8+nt]);
      v16bf b2 = load_frag_bt(whc1t, nt*16, DH, kt*32, lane);
      cc[nt]   = wmma_bf(a0, b2, cc[nt]);
      cc[8+nt] = wmma_bf(a1, b2, cc[8+nt]);
    }
  }
  int l = lane & 15, mo = (lane >> 4) ? 8 : 0;
  #pragma unroll
  for(int mt=0; mt<2; ++mt)
    #pragma unroll
    for(int nt=0; nt<8; ++nt)
      #pragma unroll
      for(int r=0; r<8; ++r){
        int row = row0 + mt*16 + r + mo;
        int qq = q0 + row;
        if(qq < NEX){
          eg [(size_t)qq*DH + nt*16 + l] = cg[mt*8+nt][r];
          ehc[(size_t)qq*DH + nt*16 + l] = cc[mt*8+nt][r];
        }
      }
  if(ok){
    float s = 0.f;
    for(int d=0; d<DH; ++d) s += bf2f(sE[tid*DH + d]) * Wp[128 + d];
    ydot[q] = s;
  }
}

// ===================== per-timestep GAT kernel =====================
struct StepArgs {
  const int* questions; const float* features; const int* adj; const int* qt_mat;
  const float* ex_emb;
  const float* gat_a1; const float* gat_a2; const float* out_a1; const float* out_a2;
  const float* Wp; const float* bp;
  const u16* wgatT; const u16* outwT; const u16* wg0T; const u16* whc0T;
  const float* rg; const float* rhc; const float* eg; const float* ehc; const float* ydot;
  float* gpre; float* hcpre; float* ypre;
};

__device__ __forceinline__ void softmax_row(const float* f1, const float* f2,
                                            const unsigned* mb, float* rmax, float* rinv, int i){
  float F1 = f1[i];
  float mx = -3.0e38f;
  for(int j=0;j<KC;++j){
    float s = F1 + f2[j]; s = (s > 0.f) ? s : 0.2f*s;
    bool mk = (mb[i*8 + (j>>5)] >> (j & 31)) & 1u;
    if(!mk) s = -3.0e38f;
    mx = fmaxf(mx, s);
  }
  float sum = 0.f;
  for(int j=0;j<KC;++j){
    float s = F1 + f2[j]; s = (s > 0.f) ? s : 0.2f*s;
    bool mk = (mb[i*8 + (j>>5)] >> (j & 31)) & 1u;
    sum += mk ? __expf(s - mx) : 0.f;
  }
  rmax[i] = mx; rinv[i] = 1.f / sum;
}

__global__ void __launch_bounds__(256) step_k(StepArgs A){
  __shared__ u16 sA[KC*DH];     // kc_avg row-major [kc][d]; later EG/EHc [kc][o]   (64 KB)
  __shared__ u16 sT[KC*DH];     // WhT[d][kc](ld256) / elu(hp)[kc][d](ld128)
                                // / Wh2T[o][kc](ld256) / enh[kc][o](ld128)          (64 KB)
  __shared__ unsigned smask[KC*8];                                              //  (8 KB)
  __shared__ float sf1[KC], sf2[KC], srmax[KC], srinv[KC], sz[KC];
  __shared__ int sq[BB], sqn[BB], sfb[BB];

  const int t = blockIdx.x;
  const int tid = threadIdx.x, lane = tid & 31, wv = tid >> 5;
  const int row0 = wv * 32;    // 32-row ownership (M=256 GEMMs)
  const int dr0  = wv * 16;    // 16-row ownership (M=128 GEMMs)

  if(tid < NH){
    __builtin_prefetch(A.wgatT + (size_t)tid*DH*DH, 0, 1);
    __builtin_prefetch(A.outwT + (size_t)tid*DH*DH, 0, 1);
  }

  sq[tid]  = A.questions[tid*TT + t];
  sqn[tid] = A.questions[tid*TT + t + 1];
  sfb[tid] = (A.features[tid*TT + t] >= 0.5f) ? 1 : 0;
  #pragma unroll
  for(int w=0; w<8; ++w){
    unsigned bits = 0;
    const int4* ap = reinterpret_cast<const int4*>(A.adj + tid*KC + w*32);
    #pragma unroll
    for(int v=0; v<8; ++v){
      int4 a4 = ap[v];
      if(a4.x) bits |= 1u << (v*4 + 0);
      if(a4.y) bits |= 1u << (v*4 + 1);
      if(a4.z) bits |= 1u << (v*4 + 2);
      if(a4.w) bits |= 1u << (v*4 + 3);
    }
    smask[tid*8 + w] = bits;
  }
  __syncthreads();

  // ---- kc_avg: wave-cooperative sparse column reduce (ballot over batch) ----
  for(int k = row0; k < row0 + 32; ++k){
    float a0=0.f, a1=0.f, a2=0.f, a3=0.f; int cnt=0;
    for(int bb=0; bb<8; ++bb){
      int b = bb*32 + lane;
      int hit = (A.qt_mat[(size_t)sq[b]*KC + k] != 0) ? 1 : 0;
      unsigned long long m = __ballot(hit);
      cnt += __popcll(m);
      while(m){
        int sb = __ffsll((unsigned long long)m) - 1; m &= (m - 1ull);
        float4 e = *reinterpret_cast<const float4*>(A.ex_emb + (size_t)sq[bb*32 + sb]*DH + lane*4);
        a0 += e.x; a1 += e.y; a2 += e.z; a3 += e.w;
      }
    }
    float inv = (cnt > 0) ? 1.f/((float)cnt + 1e-8f) : 0.f;
    u16* dst = sA + (size_t)k*DH + lane*4;
    dst[0]=f2bf(a0*inv); dst[1]=f2bf(a1*inv); dst[2]=f2bf(a2*inv); dst[3]=f2bf(a3*inv);
  }
  __syncthreads();

  // ---- layer-1 heads; Wh2^T accumulated in registers across heads ----
  v8f c2[16];
  #pragma unroll
  for(int i=0;i<16;++i) zero8(c2[i]);

  for(int hd=0; hd<NH; ++hd){
    // WhT = W^T @ kc_avg^T : M=128(dout), N=256(kc), K=128(din)
    const u16* WT = A.wgatT + (size_t)hd*DH*DH;
    v8f cw[16];
    #pragma unroll
    for(int i=0;i<16;++i) zero8(cw[i]);
    #pragma unroll
    for(int kt=0; kt<4; ++kt){
      v16bf a = load_frag_rm(WT, dr0, DH, kt*32, lane);
      #pragma unroll
      for(int nt=0; nt<16; ++nt){
        v16bf b = load_frag_bt(sA, nt*16, DH, kt*32, lane);
        cw[nt] = wmma_bf(a, b, cw[nt]);
      }
    }
    __syncthreads();                       // previous users of sT done
    #pragma unroll
    for(int nt=0; nt<16; ++nt)
      store_d_bf(sT, cw[nt], dr0, 256, nt*16, lane);   // sT = WhT [d][kc]
    __syncthreads();

    {   // f1[i] = Wh[i].a1 = sum_d WhT[d][i]*a1[d]
      const float* ga1 = A.gat_a1 + hd*DH;
      const float* ga2 = A.gat_a2 + hd*DH;
      float s1=0.f, s2=0.f;
      for(int d=0; d<DH; ++d){
        float w = bf2f(sT[(size_t)d*256 + tid]);
        s1 += w * ga1[d]; s2 += w * ga2[d];
      }
      sf1[tid]=s1; sf2[tid]=s2;
    }
    __syncthreads();
    softmax_row(sf1, sf2, smask, srmax, srinv, tid);
    __syncthreads();

    // hp = att @ Wh : M=256(kc), N=128(d), K=256(kc); B from WhT (transposed layout)
    v8f ch[16];
    #pragma unroll
    for(int i=0;i<16;++i) zero8(ch[i]);
    #pragma unroll
    for(int kt=0; kt<8; ++kt){
      v16bf a0 = att_frag(sf1, sf2, srmax, srinv, smask, row0,    kt*32, lane);
      v16bf a1 = att_frag(sf1, sf2, srmax, srinv, smask, row0+16, kt*32, lane);
      #pragma unroll
      for(int nt=0; nt<8; ++nt){
        v16bf b = load_frag_bt(sT, nt*16, 256, kt*32, lane);
        ch[nt]   = wmma_bf(a0, b, ch[nt]);
        ch[8+nt] = wmma_bf(a1, b, ch[8+nt]);
      }
    }
    __syncthreads();
    // elu, store row-major [kc][d] into sT
    #pragma unroll
    for(int i=0;i<16;++i)
      #pragma unroll
      for(int r=0;r<8;++r){
        float x = ch[i][r];
        ch[i][r] = (x > 0.f) ? x : (__expf(x) - 1.f);
      }
    #pragma unroll
    for(int mt=0; mt<2; ++mt)
      #pragma unroll
      for(int nt=0; nt<8; ++nt)
        store_d_bf(sT, ch[mt*8+nt], row0 + mt*16, DH, nt*16, lane);
    __syncthreads();
    // c2^T += outW^T @ elu(hp)^T : M=128(o), N=256(kc), K=128(d)
    const u16* OT = A.outwT + (size_t)hd*DH*DH;
    #pragma unroll
    for(int kt=0; kt<4; ++kt){
      v16bf a = load_frag_rm(OT, dr0, DH, kt*32, lane);
      #pragma unroll
      for(int nt=0; nt<16; ++nt){
        v16bf b = load_frag_bt(sT, nt*16, DH, kt*32, lane);
        c2[nt] = wmma_bf(a, b, c2[nt]);
      }
    }
  }

  // ---- layer 2 ----
  __syncthreads();
  #pragma unroll
  for(int nt=0; nt<16; ++nt)
    store_d_bf(sT, c2[nt], dr0, 256, nt*16, lane);     // sT = Wh2T [o][kc]
  __syncthreads();
  {
    float s1=0.f, s2=0.f;
    for(int d=0; d<DH; ++d){
      float w = bf2f(sT[(size_t)d*256 + tid]);
      s1 += w * A.out_a1[d]; s2 += w * A.out_a2[d];
    }
    sf1[tid]=s1; sf2[tid]=s2;
  }
  __syncthreads();
  softmax_row(sf1, sf2, smask, srmax, srinv, tid);
  __syncthreads();

  // enh = att2 @ Wh2 : M=256(kc), N=128(o), K=256(kc)
  v8f ce[16];
  #pragma unroll
  for(int i=0;i<16;++i) zero8(ce[i]);
  #pragma unroll
  for(int kt=0; kt<8; ++kt){
    v16bf a0 = att_frag(sf1, sf2, srmax, srinv, smask, row0,    kt*32, lane);
    v16bf a1 = att_frag(sf1, sf2, srmax, srinv, smask, row0+16, kt*32, lane);
    #pragma unroll
    for(int nt=0; nt<8; ++nt){
      v16bf b = load_frag_bt(sT, nt*16, 256, kt*32, lane);
      ce[nt]   = wmma_bf(a0, b, ce[nt]);
      ce[8+nt] = wmma_bf(a1, b, ce[8+nt]);
    }
  }
  __syncthreads();
  #pragma unroll
  for(int mt=0; mt<2; ++mt)
    #pragma unroll
    for(int nt=0; nt<8; ++nt)
      store_d_bf(sT, ce[mt*8+nt], row0 + mt*16, DH, nt*16, lane);   // sT = enh [kc][o]
  __syncthreads();

  {   // z[k] = enh[k] . Wp[256:384]
    float s = 0.f;
    for(int d=0; d<DH; ++d) s += bf2f(sT[tid*DH + d]) * A.Wp[256 + d];
    sz[tid] = s;
  }

  // ---- EG = enh @ Wg0 -> sA ; gpre per-b sparse sum ----
  {
    v8f cE[16];
    #pragma unroll
    for(int i=0;i<16;++i) zero8(cE[i]);
    #pragma unroll
    for(int kt=0; kt<4; ++kt){
      v16bf a0 = load_frag_rm(sT, row0,    DH, kt*32, lane);
      v16bf a1 = load_frag_rm(sT, row0+16, DH, kt*32, lane);
      #pragma unroll
      for(int nt=0; nt<8; ++nt){
        v16bf b = load_frag_bt(A.wg0T, nt*16, DH, kt*32, lane);
        cE[nt]   = wmma_bf(a0, b, cE[nt]);
        cE[8+nt] = wmma_bf(a1, b, cE[8+nt]);
      }
    }
    #pragma unroll
    for(int mt=0; mt<2; ++mt)
      #pragma unroll
      for(int nt=0; nt<8; ++nt)
        store_d_bf(sA, cE[mt*8+nt], row0 + mt*16, DH, nt*16, lane);
    __syncthreads();
    for(int b = row0; b < row0 + 32; ++b){
      int qb = sq[b];
      float a0=0.f,a1=0.f,a2=0.f,a3=0.f;
      for(int kk=0; kk<8; ++kk){
        int k = kk*32 + lane;
        int hit = (A.qt_mat[(size_t)qb*KC + k] != 0) ? 1 : 0;
        unsigned long long m = __ballot(hit);
        while(m){
          int sb = __ffsll((unsigned long long)m) - 1; m &= (m - 1ull);
          uint2 pk = *reinterpret_cast<const uint2*>(sA + (size_t)(kk*32 + sb)*DH + lane*4);
          a0 += bf2f((u16)(pk.x & 0xffff)); a1 += bf2f((u16)(pk.x >> 16));
          a2 += bf2f((u16)(pk.y & 0xffff)); a3 += bf2f((u16)(pk.y >> 16));
        }
      }
      int d0 = lane*4;
      float4 egr = *reinterpret_cast<const float4*>(A.eg + (size_t)qb*DH + d0);
      float4 rgr = *reinterpret_cast<const float4*>(A.rg + sfb[b]*DH + d0);
      float4 o;
      o.x = a0 + egr.x + rgr.x; o.y = a1 + egr.y + rgr.y;
      o.z = a2 + egr.z + rgr.z; o.w = a3 + egr.w + rgr.w;
      *reinterpret_cast<float4*>(A.gpre + ((size_t)t*BB + b)*DH + d0) = o;
    }
  }
  __syncthreads();

  // ---- EHc = enh @ Whc0 -> sA ; hcpre per-b ----
  {
    v8f cH[16];
    #pragma unroll
    for(int i=0;i<16;++i) zero8(cH[i]);
    #pragma unroll
    for(int kt=0; kt<4; ++kt){
      v16bf a0 = load_frag_rm(sT, row0,    DH, kt*32, lane);
      v16bf a1 = load_frag_rm(sT, row0+16, DH, kt*32, lane);
      #pragma unroll
      for(int nt=0; nt<8; ++nt){
        v16bf b = load_frag_bt(A.whc0T, nt*16, DH, kt*32, lane);
        cH[nt]   = wmma_bf(a0, b, cH[nt]);
        cH[8+nt] = wmma_bf(a1, b, cH[8+nt]);
      }
    }
    #pragma unroll
    for(int mt=0; mt<2; ++mt)
      #pragma unroll
      for(int nt=0; nt<8; ++nt)
        store_d_bf(sA, cH[mt*8+nt], row0 + mt*16, DH, nt*16, lane);
    __syncthreads();
    for(int b = row0; b < row0 + 32; ++b){
      int qb = sq[b];
      float a0=0.f,a1=0.f,a2=0.f,a3=0.f;
      for(int kk=0; kk<8; ++kk){
        int k = kk*32 + lane;
        int hit = (A.qt_mat[(size_t)qb*KC + k] != 0) ? 1 : 0;
        unsigned long long m = __ballot(hit);
        while(m){
          int sb = __ffsll((unsigned long long)m) - 1; m &= (m - 1ull);
          uint2 pk = *reinterpret_cast<const uint2*>(sA + (size_t)(kk*32 + sb)*DH + lane*4);
          a0 += bf2f((u16)(pk.x & 0xffff)); a1 += bf2f((u16)(pk.x >> 16));
          a2 += bf2f((u16)(pk.y & 0xffff)); a3 += bf2f((u16)(pk.y >> 16));
        }
      }
      int d0 = lane*4;
      float4 ecr = *reinterpret_cast<const float4*>(A.ehc + (size_t)qb*DH + d0);
      float4 rcr = *reinterpret_cast<const float4*>(A.rhc + sfb[b]*DH + d0);
      float4 o;
      o.x = a0 + ecr.x + rcr.x; o.y = a1 + ecr.y + rcr.y;
      o.z = a2 + ecr.z + rcr.z; o.w = a3 + ecr.w + rcr.w;
      *reinterpret_cast<float4*>(A.hcpre + ((size_t)t*BB + b)*DH + d0) = o;
    }
  }

  // ---- ypre[b] = sum_hit z[k] + ydot[q_next] + bp ----
  for(int b = row0; b < row0 + 32; ++b){
    int qb = sqn[b];
    float part = 0.f;
    for(int kk=0; kk<8; ++kk){
      int k = kk*32 + lane;
      if(A.qt_mat[(size_t)qb*KC + k] != 0) part += sz[k];
    }
    for(int off=16; off>0; off>>=1) part += __shfl_down(part, off, 32);
    if(lane == 0) A.ypre[(size_t)t*BB + b] = part + A.ydot[qb] + A.bp[0];
  }
}

// ===================== sequential recurrence =====================
struct SeqArgs {
  const u16* wghT; const u16* whchT; const float* Wp;
  const float* gpre; const float* hcpre; const float* ypre;
  float* out;
};

__global__ void __launch_bounds__(256) seq_k(SeqArgs A){
  __shared__ u16 sH[BB*DH];      // h bf16 row-major (64 KB)
  __shared__ u16 sWg[DH*DH];     // Wg[320:448]^T  [o][d] bf16 (32 KB)
  __shared__ u16 sWc[DH*DH];     // Whc[320:448]^T [o][d] bf16 (32 KB)
  __shared__ float sWp[DH];
  int tid = threadIdx.x, lane = tid & 31, wv = tid >> 5, row0 = wv*32;
  int l = lane & 15, mo = (lane >> 4) ? 8 : 0;
  for(int i=tid; i<BB*DH; i+=256) sH[i] = 0;
  for(int i=tid; i<DH*DH; i+=256){ sWg[i]=A.wghT[i]; sWc[i]=A.whchT[i]; }
  if(tid < DH) sWp[tid] = A.Wp[tid];
  v8f h[16];
  #pragma unroll
  for(int i=0;i<16;++i) zero8(h[i]);
  __syncthreads();

  for(int t=0; t<TS; ++t){
    v8f cg[16], cc[16];
    #pragma unroll
    for(int i=0;i<16;++i){ zero8(cg[i]); zero8(cc[i]); }
    #pragma unroll
    for(int kt=0; kt<4; ++kt){
      v16bf a0 = load_frag_rm(sH, row0,    DH, kt*32, lane);
      v16bf a1 = load_frag_rm(sH, row0+16, DH, kt*32, lane);
      #pragma unroll
      for(int nt=0; nt<8; ++nt){
        v16bf b1 = load_frag_bt(sWg, nt*16, DH, kt*32, lane);
        cg[nt]   = wmma_bf(a0, b1, cg[nt]);
        cg[8+nt] = wmma_bf(a1, b1, cg[8+nt]);
        v16bf b2 = load_frag_bt(sWc, nt*16, DH, kt*32, lane);
        cc[nt]   = wmma_bf(a0, b2, cc[nt]);
        cc[8+nt] = wmma_bf(a1, b2, cc[8+nt]);
      }
    }
    const float* gp = A.gpre  + (size_t)t*BB*DH;
    const float* cp = A.hcpre + (size_t)t*BB*DH;
    #pragma unroll
    for(int mt=0; mt<2; ++mt)
      #pragma unroll
      for(int nt=0; nt<8; ++nt)
        #pragma unroll
        for(int r=0; r<8; ++r){
          int row = row0 + mt*16 + r + mo, col = nt*16 + l;
          size_t idx = (size_t)row*DH + col;
          float g  = 1.f/(1.f + __expf(-(cg[mt*8+nt][r] + gp[idx])));
          float hc = tanhf(cc[mt*8+nt][r] + cp[idx]);
          h[mt*8+nt][r] = (1.f - g)*h[mt*8+nt][r] + g*hc;
        }
    __syncthreads();
    #pragma unroll
    for(int mt=0; mt<2; ++mt)
      #pragma unroll
      for(int nt=0; nt<8; ++nt)
        store_d_bf(sH, h[mt*8+nt], row0 + mt*16, DH, nt*16, lane);
    __syncthreads();
    float dot = 0.f;
    for(int d=0; d<DH; ++d) dot += bf2f(sH[tid*DH + d]) * sWp[d];
    float y = 1.f/(1.f + __expf(-(dot + A.ypre[(size_t)t*BB + tid])));
    A.out[(size_t)tid*TS + t] = y;
    __syncthreads();
  }
}

// ===================== host launch =====================
extern "C" void kernel_launch(void* const* d_in, const int* in_sizes, int n_in,
                              void* d_out, int out_size, void* d_ws, size_t ws_size,
                              hipStream_t stream){
  (void)in_sizes; (void)n_in; (void)out_size; (void)ws_size;
  const float* features = (const float*)d_in[0];
  const int*   questions= (const int*)  d_in[1];
  const int*   adj      = (const int*)  d_in[2];
  const int*   qt_mat   = (const int*)  d_in[3];
  const float* ex_emb   = (const float*)d_in[4];
  const float* resp_emb = (const float*)d_in[5];
  const float* gat_W    = (const float*)d_in[6];
  const float* gat_a1   = (const float*)d_in[7];
  const float* gat_a2   = (const float*)d_in[8];
  const float* out_W    = (const float*)d_in[9];
  const float* out_a1   = (const float*)d_in[10];
  const float* out_a2   = (const float*)d_in[11];
  const float* Wg       = (const float*)d_in[12];
  const float* bg       = (const float*)d_in[13];
  const float* Whc      = (const float*)d_in[14];
  const float* bhc      = (const float*)d_in[15];
  const float* Wp       = (const float*)d_in[16];
  const float* bp       = (const float*)d_in[17];

  char* w = (char*)d_ws; size_t off = 0;
  auto alloc = [&](size_t bytes)->char* {
    char* p = w + off; off = (off + bytes + 255) & ~(size_t)255; return p;
  };
  u16*  wgatT = (u16*)  alloc((size_t)4*DH*DH*2);
  u16*  outwT = (u16*)  alloc((size_t)4*DH*DH*2);
  u16*  wg0T  = (u16*)  alloc((size_t)DH*DH*2);
  u16*  wg1T  = (u16*)  alloc((size_t)DH*DH*2);
  u16*  wghT  = (u16*)  alloc((size_t)DH*DH*2);
  u16*  whc0T = (u16*)  alloc((size_t)DH*DH*2);
  u16*  whc1T = (u16*)  alloc((size_t)DH*DH*2);
  u16*  whchT = (u16*)  alloc((size_t)DH*DH*2);
  float* rg   = (float*)alloc((size_t)2*DH*4);
  float* rhc  = (float*)alloc((size_t)2*DH*4);
  float* ydot = (float*)alloc((size_t)NEX*4);
  float* eg   = (float*)alloc((size_t)NEX*DH*4);
  float* ehc  = (float*)alloc((size_t)NEX*DH*4);
  float* gpre = (float*)alloc((size_t)TS*BB*DH*4);
  float* hcpre= (float*)alloc((size_t)TS*BB*DH*4);
  float* ypre = (float*)alloc((size_t)TS*BB*4);

  const int NB = (DH*DH + 255)/256;   // 64 blocks per 128x128 matrix
  for(int hd=0; hd<NH; ++hd){
    cvt_bf_t_k<<<NB, 256, 0, stream>>>(gat_W + (size_t)hd*DH*DH, wgatT + (size_t)hd*DH*DH, DH*DH);
    cvt_bf_t_k<<<NB, 256, 0, stream>>>(out_W + (size_t)hd*DH*DH, outwT + (size_t)hd*DH*DH, DH*DH);
  }
  cvt_bf_t_k<<<NB, 256, 0, stream>>>(Wg,            wg0T,  DH*DH);
  cvt_bf_t_k<<<NB, 256, 0, stream>>>(Wg  + 128*DH,  wg1T,  DH*DH);
  cvt_bf_t_k<<<NB, 256, 0, stream>>>(Wg  + 320*DH,  wghT,  DH*DH);
  cvt_bf_t_k<<<NB, 256, 0, stream>>>(Whc,           whc0T, DH*DH);
  cvt_bf_t_k<<<NB, 256, 0, stream>>>(Whc + 128*DH,  whc1T, DH*DH);
  cvt_bf_t_k<<<NB, 256, 0, stream>>>(Whc + 320*DH,  whchT, DH*DH);
  resp_fold_k<<<1, 256, 0, stream>>>(resp_emb, Wg, bg, Whc, bhc, rg, rhc);
  qproj_k<<<(NEX + 255)/256, 256, 0, stream>>>(ex_emb, wg1T, whc1T, Wp, eg, ehc, ydot);

  StepArgs SA{questions, features, adj, qt_mat, ex_emb,
              gat_a1, gat_a2, out_a1, out_a2, Wp, bp,
              wgatT, outwT, wg0T, whc0T, rg, rhc, eg, ehc, ydot,
              gpre, hcpre, ypre};
  step_k<<<TS, 256, 0, stream>>>(SA);

  SeqArgs QA{wghT, whchT, Wp, gpre, hcpre, ypre, (float*)d_out};
  seq_k<<<1, 256, 0, stream>>>(QA);
}
